// CoLT5Layer_37864431681717
// MI455X (gfx1250) — matrix operand order repaired and checked by reading, not compiled, not run on hardware
//
#include <hip/hip_runtime.h>
#include <hip/hip_bf16.h>
#include <cmath>

// ---------------------------------------------------------------------------
// Types for CDNA5 WMMA (wave32)
// ---------------------------------------------------------------------------
typedef __bf16 v16bf __attribute__((ext_vector_type(16)));
typedef __bf16 v8bf  __attribute__((ext_vector_type(8)));
typedef float  v8f   __attribute__((ext_vector_type(8)));

#define BM 128
#define BN 128
#define BK 32
#define LDS_RS 40   // LDS row stride in bf16 elems (80B: 16B-aligned, conflict-breaking)

__device__ __forceinline__ float gelu_exact(float v) {
    return 0.5f * v * (1.0f + erff(v * 0.70710678118654752440f));
}

// gfx1250 async global->LDS (ASYNCcnt-tracked, no VGPR round trip, no ds_store)
__device__ __forceinline__ void async_load_b128(const __bf16* g, __bf16* l) {
    unsigned lds_addr = (unsigned)(uintptr_t)l;   // low 32 bits of generic shared ptr = LDS byte addr
    asm volatile("global_load_async_to_lds_b128 %0, %1, off"
                 :: "v"(lds_addr), "v"(g) : "memory");
}
__device__ __forceinline__ void wait_async0() {
    asm volatile("s_wait_asynccnt 0x0" ::: "memory");
}

// ---------------------------------------------------------------------------
// f32 -> bf16 elementwise conversion
// ---------------------------------------------------------------------------
__global__ void cvt_bf16_kernel(const float* __restrict__ in, __bf16* __restrict__ out, size_t n) {
    size_t i = (size_t)blockIdx.x * blockDim.x + threadIdx.x;
    size_t stride = (size_t)gridDim.x * blockDim.x;
    for (; i < n; i += stride) out[i] = (__bf16)in[i];
}

// ---------------------------------------------------------------------------
// f32 [R,C] -> bf16 [C,R] transposed conversion (batched over blockIdx.z)
// ---------------------------------------------------------------------------
__global__ void transpose_cvt_kernel(const float* __restrict__ in, __bf16* __restrict__ out,
                                     int R, int C) {
    __shared__ float tile[32][33];
    size_t boff = (size_t)blockIdx.z * (size_t)R * (size_t)C;
    const float* ip = in + boff;
    __bf16* op = out + boff;
    int c0 = blockIdx.x * 32;
    int r0 = blockIdx.y * 32;
#pragma unroll
    for (int j = threadIdx.y; j < 32; j += 8)
        tile[j][threadIdx.x] = ip[(size_t)(r0 + j) * C + c0 + threadIdx.x];
    __syncthreads();
#pragma unroll
    for (int j = threadIdx.y; j < 32; j += 8)
        op[(size_t)(c0 + j) * R + r0 + threadIdx.x] = (__bf16)tile[threadIdx.x][j];
}

// ---------------------------------------------------------------------------
// Tiled bf16 WMMA GEMM:  C[M,N] = A[M,K] @ Bt[N,K]^T  (+bias)
//   mode 0: outB[m,n] = bf16( gelu(acc + bias[n]) )
//   mode 1: outF[m,n] {=, +=} cw[m*4] * (acc + bias[n])   (cw pre-offset by expert)
// 256 threads = 8 waves; block tile 128x128; wave tile 32x64 (2x4 wmma).
// Double-buffered LDS fed by async global->LDS copies (ASYNCcnt pipeline).
// ---------------------------------------------------------------------------
__global__ __launch_bounds__(256) void gemm_bf16_wmma_kernel(
    const __bf16* __restrict__ A, const __bf16* __restrict__ Bt,
    const float* __restrict__ bias,
    __bf16* __restrict__ outB, float* __restrict__ outF,
    const float* __restrict__ cw,
    int M, int N, int K, int mode, int accumulate)
{
    __shared__ __align__(16) __bf16 As[2][BM * LDS_RS];
    __shared__ __align__(16) __bf16 Bs[2][BN * LDS_RS];
    __shared__ float cwS[BM];

    const int tid  = threadIdx.x;
    const int lane = tid & 31;
    const int wave = tid >> 5;       // 0..7
    const int wm   = wave & 3;       // row group (32 rows each) -> 128 rows
    const int wn   = wave >> 2;      // col group (64 cols each) -> 128 cols
    const int m0   = blockIdx.y * BM;
    const int n0   = blockIdx.x * BN;

    if (mode == 1 && tid < BM) cwS[tid] = cw[(size_t)(m0 + tid) * 4];

    v8f acc[2][4] = {};

    const int half = lane >> 4;      // 0 or 1
    const int l16  = lane & 15;
    const int kbA  = half * 8;       // A frag K base (lane>=16 holds K{8-15,24-31})

    const int ldr = tid >> 2;        // 0..63
    const int ldc = (tid & 3) * 8;   // 0/8/16/24 bf16 elems (16B granules)

    const __bf16* Abase = A  + (size_t)(m0 + ldr) * K + ldc;
    const __bf16* Bbase = Bt + (size_t)(n0 + ldr) * K + ldc;
    const size_t rowK64 = (size_t)64 * K;

    // stage tile kt into buffer buf (4 async b128 per thread: 128x32 A + 128x32 B)
    auto stage = [&](int kt, int buf) {
        async_load_b128(Abase + kt,          &As[buf][ldr * LDS_RS + ldc]);
        async_load_b128(Abase + kt + rowK64, &As[buf][(ldr + 64) * LDS_RS + ldc]);
        async_load_b128(Bbase + kt,          &Bs[buf][ldr * LDS_RS + ldc]);
        async_load_b128(Bbase + kt + rowK64, &Bs[buf][(ldr + 64) * LDS_RS + ldc]);
    };

    stage(0, 0);

    const int nk = K / BK;
    for (int t = 0; t < nk; t++) {
        const int cur = t & 1;
        wait_async0();           // this wave's async writes into buf[cur] complete
        __syncthreads();         // all waves' portions visible; prev reads of buf[cur^1] done
        if (t + 1 < nk) stage((t + 1) * BK, cur ^ 1);   // overlap with compute below

        // --- A fragments (16x32 bf16): lane<16 -> K{0-7,16-23}, else K{8-15,24-31}
        v16bf af[2];
#pragma unroll
        for (int i = 0; i < 2; i++) {
            const __bf16* p = &As[cur][(wm * 32 + i * 16 + l16) * LDS_RS + kbA];
            union { v16bf v; v8bf h[2]; } u;
            u.h[0] = *(const v8bf*)p;
            u.h[1] = *(const v8bf*)(p + 16);
            af[i] = u.v;
        }
        // --- B fragments (32x16 bf16): lane<16 -> K{0-7,16-23}, else K{8-15,24-31}
        v16bf bfv[4];
#pragma unroll
        for (int j = 0; j < 4; j++) {
            const __bf16* p = &Bs[cur][(wn * 64 + j * 16 + l16) * LDS_RS + half * 8];
            union { v16bf v; v8bf h[2]; } u;
            u.h[0] = *(const v8bf*)p;
            u.h[1] = *(const v8bf*)(p + 16);
            bfv[j] = u.v;
        }
#pragma unroll
        for (int i = 0; i < 2; i++)
#pragma unroll
            for (int j = 0; j < 4; j++)
                acc[i][j] = __builtin_amdgcn_wmma_f32_16x16x32_bf16(
                    false, af[i], false, bfv[j], (short)0, acc[i][j], false, false);
    }

    // --- epilogue: C/D layout: lane<16 -> n=lane, m=r ; lane>=16 -> n=lane-16, m=8+r
    const int mb = half * 8;
#pragma unroll
    for (int i = 0; i < 2; i++) {
#pragma unroll
        for (int j = 0; j < 4; j++) {
            const int gn = n0 + wn * 64 + j * 16 + l16;
            const float b = bias[gn];
#pragma unroll
            for (int r = 0; r < 8; r++) {
                const int lm = wm * 32 + i * 16 + mb + r;
                const int gm = m0 + lm;
                const float v = acc[i][j][r] + b;
                if (mode == 0) {
                    outB[(size_t)gm * N + gn] = (__bf16)gelu_exact(v);
                } else {
                    const float s = cwS[lm] * v;
                    const size_t idx = (size_t)gm * N + gn;
                    if (accumulate) outF[idx] += s;
                    else            outF[idx] = s;
                }
            }
        }
    }
}

// ---------------------------------------------------------------------------
// Router tail: logits = RH @ rW2 + rb2; softmax; top-2 -> combine weights
// One wave (32 lanes) per token. E fixed at 4.
// ---------------------------------------------------------------------------
__global__ __launch_bounds__(256) void router_topk_kernel(
    const __bf16* __restrict__ RH, const float* __restrict__ rW2,
    const float* __restrict__ rb2, float* __restrict__ cw, int Ltot, int K)
{
    const int tid   = blockIdx.x * blockDim.x + threadIdx.x;
    const int lane  = threadIdx.x & 31;
    const int token = tid >> 5;
    if (token >= Ltot) return;

    float a0 = 0.f, a1 = 0.f, a2 = 0.f, a3 = 0.f;
    const __bf16* rh = RH + (size_t)token * K;
    for (int k = lane; k < K; k += 32) {
        const float h = (float)rh[k];
        const float4 w = *(const float4*)(rW2 + (size_t)k * 4);
        a0 += h * w.x; a1 += h * w.y; a2 += h * w.z; a3 += h * w.w;
    }
#pragma unroll
    for (int off = 16; off >= 1; off >>= 1) {
        a0 += __shfl_xor(a0, off, 32);
        a1 += __shfl_xor(a1, off, 32);
        a2 += __shfl_xor(a2, off, 32);
        a3 += __shfl_xor(a3, off, 32);
    }
    if (lane == 0) {
        float l[4] = { a0 + rb2[0], a1 + rb2[1], a2 + rb2[2], a3 + rb2[3] };
        const float mx = fmaxf(fmaxf(l[0], l[1]), fmaxf(l[2], l[3]));
        float p[4]; float s = 0.f;
#pragma unroll
        for (int e = 0; e < 4; e++) { p[e] = expf(l[e] - mx); s += p[e]; }
        const float inv = 1.f / s;
#pragma unroll
        for (int e = 0; e < 4; e++) p[e] *= inv;
        int i1 = 0;
#pragma unroll
        for (int e = 1; e < 4; e++) if (p[e] > p[i1]) i1 = e;
        int i2 = -1;
#pragma unroll
        for (int e = 0; e < 4; e++) { if (e == i1) continue; if (i2 < 0 || p[e] > p[i2]) i2 = e; }
        float4 o = { 0.f, 0.f, 0.f, 0.f };
        ((float*)&o)[i1] = p[i1];
        ((float*)&o)[i2] = p[i2];
        *(float4*)(cw + (size_t)token * 4) = o;
    }
}

// ---------------------------------------------------------------------------
// Host-side orchestration
// ---------------------------------------------------------------------------
extern "C" void kernel_launch(void* const* d_in, const int* in_sizes, int n_in,
                              void* d_out, int out_size, void* d_ws, size_t ws_size,
                              hipStream_t stream) {
    const float* x   = (const float*)d_in[0];
    const float* rW1 = (const float*)d_in[1];
    const float* rb1 = (const float*)d_in[2];
    const float* rW2 = (const float*)d_in[3];
    const float* rb2 = (const float*)d_in[4];
    const float* We1 = (const float*)d_in[5];
    const float* be1 = (const float*)d_in[6];
    const float* We2 = (const float*)d_in[7];
    const float* be2 = (const float*)d_in[8];
    float* out = (float*)d_out;

    const int D  = in_sizes[2] / 2;     // rb1 has 2D elems -> D = 1024
    const int D2 = 2 * D;               // 2048
    const int M  = in_sizes[0] / D;     // B*L = 8192 tokens
    const int E  = in_sizes[4];         // 4
    const int H  = in_sizes[6] / E;     // 4096

    // workspace partition (256B aligned)
    char* ws = (char*)d_ws;
    size_t off = 0;
    auto carve = [&](size_t bytes) -> char* {
        char* p = ws + off;
        off = (off + bytes + 255) & ~(size_t)255;
        return p;
    };
    __bf16* xb    = (__bf16*)carve((size_t)M * D * 2);
    __bf16* rW1t  = (__bf16*)carve((size_t)D2 * D * 2);
    __bf16* We1t  = (__bf16*)carve((size_t)E * H * D * 2);
    __bf16* We2t  = (__bf16*)carve((size_t)E * D * H * 2);
    __bf16* RH    = (__bf16*)carve((size_t)M * D2 * 2);
    __bf16* Hbuf  = (__bf16*)carve((size_t)M * H * 2);
    float*  cw    = (float*)carve((size_t)M * E * 4);
    (void)ws_size; (void)n_in; (void)out_size;

    // 1) x -> bf16
    cvt_bf16_kernel<<<2048, 256, 0, stream>>>(x, xb, (size_t)M * D);

    // 2) transposed bf16 weights (K-contiguous rows for WMMA B fragments)
    {
        dim3 blk(32, 8);
        transpose_cvt_kernel<<<dim3(D2 / 32, D / 32, 1), blk, 0, stream>>>(rW1, rW1t, D, D2);
        transpose_cvt_kernel<<<dim3(H / 32, D / 32, E), blk, 0, stream>>>(We1, We1t, D, H);
        transpose_cvt_kernel<<<dim3(D / 32, H / 32, E), blk, 0, stream>>>(We2, We2t, H, D);
    }

    // 3) router GEMM1: RH = gelu(x @ rW1 + rb1)   [M, 2D]
    gemm_bf16_wmma_kernel<<<dim3(D2 / BN, M / BM), 256, 0, stream>>>(
        xb, rW1t, rb1, RH, nullptr, nullptr, M, D2, D, 0, 0);

    // 4) router GEMM2 + softmax + top-2 -> combine weights [M, E]
    router_topk_kernel<<<(M * 32 + 255) / 256, 256, 0, stream>>>(RH, rW2, rb2, cw, M, D2);

    // 5) experts: out = sum_e cw[:,e] * (gelu(x @ We1[e] + be1[e]) @ We2[e] + be2[e])
    for (int e = 0; e < E; e++) {
        gemm_bf16_wmma_kernel<<<dim3(H / BN, M / BM), 256, 0, stream>>>(
            xb, We1t + (size_t)e * H * D, be1 + (size_t)e * H,
            Hbuf, nullptr, nullptr, M, H, D, 0, 0);
        gemm_bf16_wmma_kernel<<<dim3(D / BN, M / BM), 256, 0, stream>>>(
            Hbuf, We2t + (size_t)e * D * H, be2 + (size_t)e * D,
            nullptr, out, cw + e, M, D, H, 1, e > 0 ? 1 : 0);
    }
}
